// SelfMultiheadAttn_60146722013486
// MI455X (gfx1250) — matrix operand order repaired and checked by reading, not compiled, not run on hardware
//
#include <hip/hip_runtime.h>
#include <hip/hip_bf16.h>

// ---------------------------------------------------------------------------
// Self multi-head attention fwd for MI455X (gfx1250), bf16 WMMA + fp32 accum.
// T=2048, B=4, E=1024, H=16, D=64. wave32; WMMA 16x16x32 bf16.
// GEMMs: double-buffered LDS tiles fed by GLOBAL_LOAD_ASYNC_TO_LDS_B128.
// ---------------------------------------------------------------------------

typedef __bf16 bf16_t;
typedef __attribute__((ext_vector_type(16))) __bf16 v16bf;
typedef __attribute__((ext_vector_type(8)))  float  v8f;
typedef __attribute__((ext_vector_type(4)))  int    v4i;

#define T_SEQ   2048
#define BATCH   4
#define EMB     1024
#define NHEAD   16
#define HDIM    64
#define M_ROWS  (T_SEQ * BATCH)   // 8192

#if defined(__AMDGCN__) && __has_builtin(__builtin_amdgcn_global_load_async_to_lds_b128)
#define HAS_ASYNC_LDS 1
#else
#define HAS_ASYNC_LDS 0
#endif

#if HAS_ASYNC_LDS
typedef __attribute__((address_space(1))) v4i* gv4i_p;   // global v4i*
typedef __attribute__((address_space(3))) v4i* lv4i_p;   // LDS v4i*
#endif

// 16-byte global -> LDS copy (async on CDNA5 if the builtin is available).
__device__ inline void cp16_g2l(const bf16_t* g, bf16_t* l) {
#if HAS_ASYNC_LDS
  __builtin_amdgcn_global_load_async_to_lds_b128(
      (gv4i_p)g, (lv4i_p)l, /*offset=*/0, /*cpol=*/0);
#else
  *(uint4*)l = *(const uint4*)g;
#endif
}

__device__ inline void async_wait_all() {
#if HAS_ASYNC_LDS
  asm volatile("s_wait_asynccnt 0x0" ::: "memory");
#endif
}

union FragU { v16bf v; uint4 q[2]; };

__device__ inline v8f v8f_zero() {
  v8f z;
#pragma unroll
  for (int i = 0; i < 8; ++i) z[i] = 0.0f;
  return z;
}

// WMMA D = A(16x32 bf16) * B(32x16 bf16) + C(16x16 f32)
__device__ inline v8f wmma_bf16(v16bf a, v16bf b, v8f c) {
  return __builtin_amdgcn_wmma_f32_16x16x32_bf16(
      /*neg_a=*/false, a, /*neg_b=*/false, b,
      /*c_mod=*/(short)0, c, /*reuse_a=*/false, /*reuse_b=*/false);
}

// A-fragment: 16(M) x 32(K) from row-major [M][K] (ld in elements).
// ISA layout: lanes 0-15 (M=lane): elems 0-7 -> K0-7, elems 8-15 -> K16-23
//             lanes 16-31 (M=lane-16): elems 0-7 -> K8-15, 8-15 -> K24-31
__device__ inline v16bf load_a_frag(const bf16_t* p, int ld) {
  const int lane = threadIdx.x & 31;
  const int m    = lane & 15;
  const int kb   = (lane >> 4) << 3;   // 0 or 8
  FragU f;
  f.q[0] = *(const uint4*)(p + m * ld + kb);
  f.q[1] = *(const uint4*)(p + m * ld + 16 + kb);
  return f.v;
}

// B-fragment: 32(K) x 16(N), source stored transposed [N][K] row-major.
// ISA layout: lanes 0-15 (N=lane): elems 0-7 -> K0-7, 8-15 -> K16-23; lanes 16-31: K8-15, K24-31
__device__ inline v16bf load_b_frag_nk(const bf16_t* p, int ld) {
  const int lane = threadIdx.x & 31;
  const int n    = lane & 15;
  const int kb   = (lane >> 4) << 3;   // 0 or 8
  FragU f;
  f.q[0] = *(const uint4*)(p + n * ld + kb);
  f.q[1] = *(const uint4*)(p + n * ld + 16 + kb);
  return f.v;
}

// ---------------------------------------------------------------------------
// fp32 -> bf16 conversion
// ---------------------------------------------------------------------------
__global__ __launch_bounds__(256) void f32_to_bf16_kernel(
    const float* __restrict__ in, bf16_t* __restrict__ out, int n) {
  int i = blockIdx.x * 256 + threadIdx.x;
  if (i < n) out[i] = (bf16_t)in[i];
}

// ---------------------------------------------------------------------------
// NT GEMM: C[m,n] = sum_k A[m,k] * B[n,k]   (A:[M,K] bf16, B:[N,K] bf16)
// Block tile 128x128x32, 256 threads = 8 waves, each wave 2x4 WMMA tiles.
// Double-buffered LDS, async global->LDS copies overlap WMMA compute.
// EPI==0: plain f32 store to Cout[M,N]
// EPI==1: scatter to Q/K/V [B,H,T,D] bf16, Q scaled by D^-0.5
// ---------------------------------------------------------------------------
#define LDK 40   // padded LDS row stride (elements) for 32-wide K tiles

template <int EPI>
__global__ __launch_bounds__(256) void gemm_nt_kernel(
    const bf16_t* __restrict__ A, const bf16_t* __restrict__ B,
    float* __restrict__ Cout,
    bf16_t* __restrict__ Qo, bf16_t* __restrict__ Ko, bf16_t* __restrict__ Vo,
    int Ndim, int Kdim) {
  __shared__ bf16_t lA[2][128 * LDK];
  __shared__ bf16_t lB[2][128 * LDK];

  const int bm  = blockIdx.y * 128;
  const int bn  = blockIdx.x * 128;
  const int tid = threadIdx.x;
  const int w   = tid >> 5;
  const int mt0 = (w & 3) * 2;   // wave owns m-tiles mt0, mt0+1
  const int nt0 = (w >> 2) * 4;  // wave owns n-tiles nt0..nt0+3

  // this thread's two 16B chunks per 128x32 tile
  const int row0 = tid >> 2;
  const int col0 = (tid & 3) << 3;
  const int row1 = (256 + tid) >> 2;
  const int col1 = col0;

  auto issue_tile = [&](int k0, int bufi) {
    cp16_g2l(&A[(size_t)(bm + row0) * Kdim + k0 + col0], &lA[bufi][row0 * LDK + col0]);
    cp16_g2l(&B[(size_t)(bn + row0) * Kdim + k0 + col0], &lB[bufi][row0 * LDK + col0]);
    cp16_g2l(&A[(size_t)(bm + row1) * Kdim + k0 + col1], &lA[bufi][row1 * LDK + col1]);
    cp16_g2l(&B[(size_t)(bn + row1) * Kdim + k0 + col1], &lB[bufi][row1 * LDK + col1]);
  };

  v8f acc[2][4];
#pragma unroll
  for (int i = 0; i < 2; ++i)
#pragma unroll
    for (int j = 0; j < 4; ++j) acc[i][j] = v8f_zero();

  issue_tile(0, 0);
  async_wait_all();
  __syncthreads();

  const int nk = Kdim >> 5;
  for (int kt = 0; kt < nk; ++kt) {
    const int buf = kt & 1;
    if (kt + 1 < nk) issue_tile((kt + 1) << 5, buf ^ 1);  // prefetch next tile

    v16bf af[2], bq[4];
#pragma unroll
    for (int i = 0; i < 2; ++i)
      af[i] = load_a_frag(&lA[buf][(mt0 + i) * 16 * LDK], LDK);
#pragma unroll
    for (int j = 0; j < 4; ++j)
      bq[j] = load_b_frag_nk(&lB[buf][(nt0 + j) * 16 * LDK], LDK);
#pragma unroll
    for (int i = 0; i < 2; ++i)
#pragma unroll
      for (int j = 0; j < 4; ++j) acc[i][j] = wmma_bf16(af[i], bq[j], acc[i][j]);

    async_wait_all();   // next tile's async copies landed
    __syncthreads();    // everyone done reading buf before it is overwritten
  }

  // epilogue (C layout: col = lane&15, rows = (lane>>4)*8 + r)
  const int lane  = tid & 31;
  const int nlane = lane & 15;
  const int mbase = (lane >> 4) * 8;
#pragma unroll
  for (int i = 0; i < 2; ++i) {
#pragma unroll
    for (int j = 0; j < 4; ++j) {
#pragma unroll
      for (int r = 0; r < 8; ++r) {
        const int m = bm + (mt0 + i) * 16 + mbase + r;
        const int n = bn + (nt0 + j) * 16 + nlane;
        const float v = acc[i][j][r];
        if (EPI == 0) {
          Cout[(size_t)m * Ndim + n] = v;
        } else {
          const int t = m >> 2, b = m & 3;      // m = t*B + b, B=4
          const int which = n >> 10;            // 0=q 1=k 2=v
          const int e = n & 1023;
          const int h = e >> 6, d = e & 63;
          const size_t dst = ((size_t)(b * NHEAD + h) * T_SEQ + t) * HDIM + d;
          bf16_t bv = (bf16_t)(which == 0 ? v * 0.125f : v);  // D^-0.5 = 1/8
          bf16_t* dp = (which == 0) ? Qo : (which == 1) ? Ko : Vo;
          dp[dst] = bv;
        }
      }
    }
  }
}

// ---------------------------------------------------------------------------
// Flash attention (causal). Block = one (b,h) x 128 query rows, 8 waves.
// Key tiles of 32; K tile async-copied into LDS [key][d], V staged
// transposed [d][key] so PV B-fragments are contiguous b128 reads.
// ---------------------------------------------------------------------------
#define LDKK 72  // K tile row stride (64 d + pad)
#define LDVT 40  // Vt / P row stride (32 + pad)

__global__ __launch_bounds__(256) void flash_attn_kernel(
    const bf16_t* __restrict__ Q, const bf16_t* __restrict__ K,
    const bf16_t* __restrict__ V, bf16_t* __restrict__ Ctx) {
  __shared__ bf16_t lK[32 * LDKK];       // [key][d]
  __shared__ bf16_t lVt[64 * LDVT];      // [d][key] transposed
  __shared__ bf16_t lP[8][16 * LDVT];    // per-wave P staging [row][key]

  const int bh   = blockIdx.x;           // b*H + h
  const int qt   = blockIdx.y;           // 128-row query tile
  const int tid  = threadIdx.x;
  const int w    = tid >> 5;
  const int lane = tid & 31;
  const int nlane = lane & 15;
  const int mbase = (lane >> 4) * 8;
  const int qb    = qt * 128 + w * 16;   // wave's first query row

  // Q fragments held in registers for the whole block
  const size_t baseQ = ((size_t)bh * T_SEQ + qb) * HDIM;
  const v16bf aq0 = load_a_frag(Q + baseQ + 0, HDIM);
  const v16bf aq1 = load_a_frag(Q + baseQ + 32, HDIM);

  v8f o[4];
#pragma unroll
  for (int j = 0; j < 4; ++j) o[j] = v8f_zero();
  float rowm[8], rowl[8];
#pragma unroll
  for (int r = 0; r < 8; ++r) { rowm[r] = -3.0e38f; rowl[r] = 0.0f; }

  const int nkt = (qt + 1) * 4;          // causal: keys < (qt+1)*128

  for (int kt = 0; kt < nkt; ++kt) {
    __syncthreads();
    {  // stage K (async, 16B/thread) and V transposed
      const int krow = tid >> 3;
      const int dcol = (tid & 7) << 3;
      const size_t gbase = ((size_t)bh * T_SEQ + kt * 32 + krow) * HDIM + dcol;
      cp16_g2l(&K[gbase], &lK[krow * LDKK + dcol]);
      uint4 vv = *(const uint4*)&V[gbase];
      const bf16_t* pv = (const bf16_t*)&vv;
#pragma unroll
      for (int e = 0; e < 8; ++e) lVt[(dcol + e) * LDVT + krow] = pv[e];
    }
    async_wait_all();
    __syncthreads();

    // S = Q K^T  (two 16x16 fragments over the 32-key tile; contract d=64)
    v8f s[2];
#pragma unroll
    for (int j = 0; j < 2; ++j) {
      v8f a = v8f_zero();
      a = wmma_bf16(aq0, load_b_frag_nk(&lK[(j * 16) * LDKK + 0],  LDKK), a);
      a = wmma_bf16(aq1, load_b_frag_nk(&lK[(j * 16) * LDKK + 32], LDKK), a);
      s[j] = a;
    }

    // causal mask + online softmax (row stats across the 16-lane half-wave)
    float mnew[8];
#pragma unroll
    for (int r = 0; r < 8; ++r) {
      const int row = qb + mbase + r;
#pragma unroll
      for (int j = 0; j < 2; ++j) {
        const int key = kt * 32 + j * 16 + nlane;
        if (key > row) s[j][r] = -3.0e38f;
      }
      float t = fmaxf(s[0][r], s[1][r]);
#pragma unroll
      for (int off = 8; off >= 1; off >>= 1) t = fmaxf(t, __shfl_xor(t, off, 32));
      mnew[r] = fmaxf(rowm[r], t);
    }
#pragma unroll
    for (int r = 0; r < 8; ++r) {
      const float scale = __expf(rowm[r] - mnew[r]);
      const float p0 = __expf(s[0][r] - mnew[r]);
      const float p1 = __expf(s[1][r] - mnew[r]);
      float psum = p0 + p1;
#pragma unroll
      for (int off = 8; off >= 1; off >>= 1) psum += __shfl_xor(psum, off, 32);
      rowl[r] = rowl[r] * scale + psum;
      rowm[r] = mnew[r];
#pragma unroll
      for (int j = 0; j < 4; ++j) o[j][r] *= scale;
      lP[w][(mbase + r) * LDVT + nlane]      = (bf16_t)p0;
      lP[w][(mbase + r) * LDVT + 16 + nlane] = (bf16_t)p1;
    }

    // O += P @ V  (contract 32 keys; 4 d-tiles of 16)
    const v16bf ap = load_a_frag(&lP[w][0], LDVT);
#pragma unroll
    for (int j = 0; j < 4; ++j)
      o[j] = wmma_bf16(ap, load_b_frag_nk(&lVt[(j * 16) * LDVT], LDVT), o[j]);
  }

  // write ctx as [t*B + b][E] bf16 for the out-projection GEMM
  const int b = bh / NHEAD, h = bh % NHEAD;
#pragma unroll
  for (int j = 0; j < 4; ++j) {
#pragma unroll
    for (int r = 0; r < 8; ++r) {
      const int t = qb + mbase + r;
      const size_t dst = ((size_t)t * BATCH + b) * EMB + h * HDIM + j * 16 + nlane;
      Ctx[dst] = (bf16_t)(o[j][r] / rowl[r]);
    }
  }
}

// ---------------------------------------------------------------------------
extern "C" void kernel_launch(void* const* d_in, const int* in_sizes, int n_in,
                              void* d_out, int out_size, void* d_ws, size_t ws_size,
                              hipStream_t stream) {
  const float* q_in  = (const float*)d_in[0];  // [T,B,E]
  const float* w_in  = (const float*)d_in[1];  // [3E,E]
  const float* w_out = (const float*)d_in[2];  // [E,E]
  float* out = (float*)d_out;

  char* ws = (char*)d_ws;
  size_t off = 0;
  auto wsalloc = [&](size_t bytes) -> void* {
    void* p = ws + off;
    off += (bytes + 255) & ~(size_t)255;
    return p;
  };
  const size_t headElems = (size_t)BATCH * NHEAD * T_SEQ * HDIM;  // 8.4M
  bf16_t* Xbf    = (bf16_t*)wsalloc((size_t)M_ROWS * EMB * 2);
  bf16_t* Winbf  = (bf16_t*)wsalloc((size_t)3 * EMB * EMB * 2);
  bf16_t* Woutbf = (bf16_t*)wsalloc((size_t)EMB * EMB * 2);
  bf16_t* Qbf    = (bf16_t*)wsalloc(headElems * 2);
  bf16_t* Kbf    = (bf16_t*)wsalloc(headElems * 2);
  bf16_t* Vbf    = (bf16_t*)wsalloc(headElems * 2);
  bf16_t* Ctxbf  = (bf16_t*)wsalloc((size_t)M_ROWS * EMB * 2);

  // 1) fp32 -> bf16
  const int nX = M_ROWS * EMB, nWi = 3 * EMB * EMB, nWo = EMB * EMB;
  f32_to_bf16_kernel<<<(nX  + 255) / 256, 256, 0, stream>>>(q_in,  Xbf,    nX);
  f32_to_bf16_kernel<<<(nWi + 255) / 256, 256, 0, stream>>>(w_in,  Winbf,  nWi);
  f32_to_bf16_kernel<<<(nWo + 255) / 256, 256, 0, stream>>>(w_out, Woutbf, nWo);

  // 2) in-projection GEMM -> Q/K/V [B,H,T,D] (Q pre-scaled)
  dim3 g1(3 * EMB / 128, M_ROWS / 128);
  gemm_nt_kernel<1><<<g1, 256, 0, stream>>>(Xbf, Winbf, nullptr,
                                            Qbf, Kbf, Vbf, 3 * EMB, EMB);

  // 3) causal flash attention -> ctx [T*B, E] bf16
  dim3 g2(BATCH * NHEAD, T_SEQ / 128);
  flash_attn_kernel<<<g2, 256, 0, stream>>>(Qbf, Kbf, Vbf, Ctxbf);

  // 4) out-projection GEMM -> fp32 output [T,B,E]
  dim3 g3(EMB / 128, M_ROWS / 128);
  gemm_nt_kernel<0><<<g3, 256, 0, stream>>>(Ctxbf, Woutbf, out,
                                            nullptr, nullptr, nullptr, EMB, EMB);
}
